// CSI_75453985457421
// MI455X (gfx1250) — matrix operand order repaired and checked by reading, not compiled, not run on hardware
//
#include <hip/hip_runtime.h>
#include <hip/hip_bf16.h>
#include <math.h>

// ---------------------------------------------------------------------------
// Types for CDNA5 WMMA
// ---------------------------------------------------------------------------
typedef __attribute__((ext_vector_type(16))) __bf16 v16bf;
typedef __attribute__((ext_vector_type(8)))  float  v8f;

// ---------------------------------------------------------------------------
// Helpers
// ---------------------------------------------------------------------------
__device__ __forceinline__ unsigned short f2bf(float f) {
    unsigned int u = __float_as_uint(f);
    unsigned int r = u + 0x7FFFu + ((u >> 16) & 1u);   // round-to-nearest-even
    return (unsigned short)(r >> 16);
}
__device__ __forceinline__ float bf2f(unsigned short h) {
    return __uint_as_float(((unsigned int)h) << 16);
}
__device__ __forceinline__ float sigmoidf_(float x) { return 1.0f / (1.0f + expf(-x)); }
__device__ __forceinline__ float siluf_(float x)    { return x * sigmoidf_(x); }
__device__ __forceinline__ float softplusf_(float x){ return (x > 20.f) ? x : log1pf(expf(x)); }

// problem constants
#define NB    4
#define NC    256
#define NSPAT 4096            // H*W
#define DD    64              // D
#define DI    128
#define DS    16
#define MROWS (16 * 4096)     // 16 sequences * N
#define R2    (4 * 4096)      // b * N rows for outc

// ---------------------------------------------------------------------------
// Weight prep kernels (fp32 -> bf16 [+pad / +transpose])
// ---------------------------------------------------------------------------
__global__ void cvt_bf16_k(const float* __restrict__ in, unsigned short* __restrict__ out, int n) {
    int i = blockIdx.x * 256 + threadIdx.x;
    if (i < n) out[i] = f2bf(in[i]);
}
__global__ void pad_xproj_k(const float* __restrict__ in, unsigned short* __restrict__ out) {
    int i = blockIdx.x * 256 + threadIdx.x;          // 128*48
    if (i < 128 * 48) {
        int r = i / 48, c = i % 48;
        out[i] = (c < 36) ? f2bf(in[r * 36 + c]) : (unsigned short)0;
    }
}
__global__ void transpose_cvt_k(const float* __restrict__ in, unsigned short* __restrict__ out) {
    int i = blockIdx.x * 256 + threadIdx.x;          // 256*256, in[o*256+c]
    if (i < 256 * 256) {
        int o = i >> 8, c = i & 255;
        out[c * 256 + o] = f2bf(in[i]);
    }
}

// ---------------------------------------------------------------------------
// Input LayerNorm over C=256, emit channel-chunk layout (16, N, 64) f32 + bf16
// ---------------------------------------------------------------------------
__global__ __launch_bounds__(256) void ln_in_k(const float* __restrict__ x,
                                               const float* __restrict__ g,
                                               const float* __restrict__ b,
                                               float* __restrict__ ch_f32,
                                               unsigned short* __restrict__ ch_bf) {
    __shared__ float red[256];
    int row = blockIdx.x;                 // bb*4096 + n
    int bb  = row >> 12;
    int n   = row & 4095;
    int t   = threadIdx.x;                // channel c
    float v = x[((size_t)bb * NC + t) * NSPAT + n];

    red[t] = v; __syncthreads();
    for (int s = 128; s > 0; s >>= 1) { if (t < s) red[t] += red[t + s]; __syncthreads(); }
    float mean = red[0] * (1.0f / 256.0f);
    __syncthreads();
    float d = v - mean;
    red[t] = d * d; __syncthreads();
    for (int s = 128; s > 0; s >>= 1) { if (t < s) red[t] += red[t + s]; __syncthreads(); }
    float var = red[0] * (1.0f / 256.0f);

    float xn = d * rsqrtf(var + 1e-5f) * g[t] + b[t];
    int chunk = t >> 6, dloc = t & 63;
    int s = chunk * NB + bb;              // sequence index 0..15
    size_t o = ((size_t)s * NSPAT + n) * DD + dloc;
    ch_f32[o] = xn;
    ch_bf[o]  = f2bf(xn);
}

// ---------------------------------------------------------------------------
// Depthwise causal conv (k=4) + SiLU.  Reads xc half of xz (f32), writes bf16.
// ---------------------------------------------------------------------------
__global__ __launch_bounds__(256) void conv_silu_k(const float* __restrict__ xz,
                                                   const float* __restrict__ cw,
                                                   const float* __restrict__ cb,
                                                   unsigned short* __restrict__ xc_bf) {
    size_t idx = (size_t)blockIdx.x * 256 + threadIdx.x;   // 16*4096*128
    int ch = (int)(idx & 127);
    int n  = (int)((idx >> 7) & 4095);
    int s  = (int)(idx >> 19);
    float acc = cb[ch];
#pragma unroll
    for (int j = 0; j < 4; ++j) {
        int nn = n - 3 + j;
        if (nn >= 0) acc += cw[ch * 4 + j] * xz[((size_t)s * NSPAT + nn) * 256 + ch];
    }
    xc_bf[idx] = f2bf(siluf_(acc));
}

// ---------------------------------------------------------------------------
// Fused selective scan: dt-proj + softplus + recurrence + D skip + silu(z) gate
// One block per sequence (128 threads = 4 waves), h[16] in registers.
// Per-step rows of xdbl (48 floats: dt[4], B[16], C[16], pad) staged in LDS.
// ---------------------------------------------------------------------------
__global__ __launch_bounds__(128) void scan_k(const float* __restrict__ xdbl,       // 65536 x 48
                                              const unsigned short* __restrict__ xc_bf,
                                              const float* __restrict__ xz,         // z at col 128..255
                                              const float* __restrict__ dtw,        // 4 x 128
                                              const float* __restrict__ dtb,        // 128
                                              const float* __restrict__ A_log,      // 128 x 16
                                              const float* __restrict__ Dp,         // 128
                                              unsigned short* __restrict__ yz) {    // 65536 x 128
    __shared__ float sh[128 * 48];        // 24 KB
    int s = blockIdx.x;
    int t = threadIdx.x;                  // channel

    float a[16];
#pragma unroll
    for (int i = 0; i < 16; ++i) a[i] = -expf(A_log[t * 16 + i]);
    float w0 = dtw[0 * 128 + t], w1 = dtw[1 * 128 + t];
    float w2 = dtw[2 * 128 + t], w3 = dtw[3 * 128 + t];
    float db = dtb[t], dpar = Dp[t];

    float h[16];
#pragma unroll
    for (int i = 0; i < 16; ++i) h[i] = 0.f;

    for (int n0 = 0; n0 < NSPAT; n0 += 128) {
        const float* src = xdbl + ((size_t)s * NSPAT + n0) * 48;   // contiguous 6144 floats
        for (int i = t; i < 128 * 48; i += 128) sh[i] = src[i];
        __syncthreads();

        for (int nl = 0; nl < 128; ++nl) {
            const float* row = sh + nl * 48;
            float dt = softplusf_(row[0] * w0 + row[1] * w1 + row[2] * w2 + row[3] * w3 + db);
            size_t gidx = (size_t)s * NSPAT + n0 + nl;
            float u  = bf2f(xc_bf[gidx * 128 + t]);
            float du = dt * u;
            float y = 0.f;
#pragma unroll
            for (int i = 0; i < 16; ++i) {
                float e = __expf(dt * a[i]);
                h[i] = e * h[i] + du * row[4 + i];
                y += h[i] * row[20 + i];
            }
            y += u * dpar;
            float z = xz[gidx * 256 + 128 + t];
            yz[gidx * 128 + t] = f2bf(y * siluf_(z));
        }
        __syncthreads();
    }
}

// ---------------------------------------------------------------------------
// LayerNorm over D=64 (wave-per-row, 2 elems/lane), f32 in -> bf16 out
// ---------------------------------------------------------------------------
__global__ __launch_bounds__(256) void ln_d64_k(const float* __restrict__ m,
                                                const float* __restrict__ g,
                                                const float* __restrict__ b,
                                                unsigned short* __restrict__ out) {
    int lane = threadIdx.x & 31;
    int wv   = threadIdx.x >> 5;
    size_t row = (size_t)blockIdx.x * 8 + wv;
    const float* p = m + row * DD;
    float v0 = p[lane], v1 = p[lane + 32];
    float sum = v0 + v1;
#pragma unroll
    for (int mk = 16; mk > 0; mk >>= 1) sum += __shfl_xor(sum, mk, 32);
    float mean = sum * (1.0f / 64.0f);
    float d0 = v0 - mean, d1 = v1 - mean;
    float sq = d0 * d0 + d1 * d1;
#pragma unroll
    for (int mk = 16; mk > 0; mk >>= 1) sq += __shfl_xor(sq, mk, 32);
    float rs = rsqrtf(sq * (1.0f / 64.0f) + 1e-5f);
    out[row * DD + lane]      = f2bf(d0 * rs * g[lane]      + b[lane]);
    out[row * DD + lane + 32] = f2bf(d1 * rs * g[lane + 32] + b[lane + 32]);
}

// ---------------------------------------------------------------------------
// GEMM epilogues
// ---------------------------------------------------------------------------
struct EpiF32 {
    float* C; int ldc;
    __device__ void operator()(int r, int c, float v) const { C[(size_t)r * ldc + c] = v; }
};
struct EpiGelu {   // + bias, exact GELU, bf16 store
    const float* bias; unsigned short* out; int ldc;
    __device__ void operator()(int r, int c, float v) const {
        v += bias[c];
        float ge = 0.5f * v * (1.0f + erff(v * 0.70710678118654752f));
        out[(size_t)r * ldc + c] = f2bf(ge);
    }
};
struct EpiSkip {   // + bias + skip_scale*ch, scatter to xo layout (b, n, C) bf16
    const float* bias; const float* skip; const float* scale; unsigned short* xo;
    __device__ void operator()(int r, int c, float v) const {
        v += bias[c] + scale[0] * skip[(size_t)r * DD + c];
        int s = r >> 12, n = r & 4095;
        int chunk = s >> 2, bb = s & 3;
        xo[((size_t)bb * NSPAT + n) * NC + c * 4 + chunk] = f2bf(v);
    }
};
struct EpiBnSilu { // batchnorm + silu, scatter to (b, o, h, w) f32
    const float* bm; const float* bv; const float* bg; const float* bb2; float* out;
    __device__ void operator()(int r, int o, float v) const {
        float t = (v - bm[o]) * rsqrtf(bv[o] + 1e-5f) * bg[o] + bb2[o];
        int bb = r >> 12, n = r & 4095;
        out[((size_t)bb * NC + o) * NSPAT + n] = siluf_(t);
    }
};

// ---------------------------------------------------------------------------
// bf16 WMMA GEMM:  C[M,N] = A[M,K] * B[K,N]
// Block: 256 threads = 8 waves; tile 128(M) x 16(N); K step 32.
// grid = (N/16, M/128).  A,B row-major bf16 (as ushort).
// ---------------------------------------------------------------------------
template <typename Epi>
__global__ __launch_bounds__(256) void gemm_bf16_k(const unsigned short* __restrict__ A,
                                                   const unsigned short* __restrict__ Bw,
                                                   int K, int lda, int ldb, Epi epi) {
    __shared__ unsigned short As[128 * 34];   // +2 pad breaks bank conflicts
    __shared__ unsigned short Bs[16 * 34];
    unsigned int* As32 = reinterpret_cast<unsigned int*>(As);
    unsigned int* Bs32 = reinterpret_cast<unsigned int*>(Bs);

    const int t       = threadIdx.x;
    const int lane    = t & 31;
    const int wv      = t >> 5;
    const int half    = lane >> 4;
    const int mloc    = lane & 15;
    const int rowBase = blockIdx.y * 128;
    const int colBase = blockIdx.x * 16;

    const int arow  = t >> 1;             // 0..127
    const int acolh = (t & 1) * 16;       // 0 or 16
    const int kB    = t >> 4;             // 0..15  (B-tile: k row)
    const int nB    = t & 15;             //        (B-tile: n col)

    v8f acc = {0.f, 0.f, 0.f, 0.f, 0.f, 0.f, 0.f, 0.f};

    for (int kk = 0; kk < K; kk += 32) {
        // --- stage A tile (128 x 32) : one 32B vector pair per thread ---
        {
            const uint4* ap = reinterpret_cast<const uint4*>(
                A + (size_t)(rowBase + arow) * lda + kk + acolh);
            uint4 q0 = ap[0];
            uint4 q1 = ap[1];
            int base = arow * 17 + (acolh >> 1);
            As32[base + 0] = q0.x; As32[base + 1] = q0.y;
            As32[base + 2] = q0.z; As32[base + 3] = q0.w;
            As32[base + 4] = q1.x; As32[base + 5] = q1.y;
            As32[base + 6] = q1.z; As32[base + 7] = q1.w;
        }
        // --- stage B tile transposed Bs[n][k] (32x16 source), fully static ---
        {
            unsigned short b0 = Bw[(size_t)(kk + kB)      * ldb + colBase + nB];
            unsigned short b1 = Bw[(size_t)(kk + kB + 16) * ldb + colBase + nB];
            Bs[nB * 34 + kB]      = b0;
            Bs[nB * 34 + kB + 16] = b1;
        }
        __syncthreads();

        // --- build fragments & WMMA ---
        union { v16bf v; unsigned int u[8]; } af, bf;
#pragma unroll
        for (int v = 0; v < 8; ++v) {
            int k0a = (v & 3) * 2 + half * 8 + (v >> 2) * 16;   // A 16x32 lane layout
            int k0b = (v & 3) * 2 + half * 8 + (v >> 2) * 16;   // B 32x16 lane layout
            af.u[v] = As32[(wv * 16 + mloc) * 17 + (k0a >> 1)];
            bf.u[v] = Bs32[mloc * 17 + (k0b >> 1)];
        }
        acc = __builtin_amdgcn_wmma_f32_16x16x32_bf16(
            false, af.v, false, bf.v, (short)0, acc, false, false);
        __syncthreads();
    }

    // --- epilogue: 16x16 f32 D layout ---
#pragma unroll
    for (int d = 0; d < 8; ++d) {
        int r = rowBase + wv * 16 + half * 8 + d;
        int c = colBase + mloc;
        epi(r, c, acc[d]);
    }
}

// ---------------------------------------------------------------------------
// Launch
// ---------------------------------------------------------------------------
extern "C" void kernel_launch(void* const* d_in, const int* in_sizes, int n_in,
                              void* d_out, int out_size, void* d_ws, size_t ws_size,
                              hipStream_t stream) {
    (void)in_sizes; (void)n_in; (void)out_size; (void)ws_size;
    const float* x        = (const float*)d_in[0];
    const float* ln_g     = (const float*)d_in[1];
    const float* ln_b     = (const float*)d_in[2];
    const float* ln1_g    = (const float*)d_in[3];
    const float* ln1_b    = (const float*)d_in[4];
    const float* in_pw    = (const float*)d_in[5];
    const float* conv_w   = (const float*)d_in[6];
    const float* conv_b   = (const float*)d_in[7];
    const float* x_pw     = (const float*)d_in[8];
    const float* dt_pw    = (const float*)d_in[9];
    const float* dt_pb    = (const float*)d_in[10];
    const float* A_log    = (const float*)d_in[11];
    const float* Dparam   = (const float*)d_in[12];
    const float* out_pw   = (const float*)d_in[13];
    const float* skip_sc  = (const float*)d_in[14];
    const float* fc1_w    = (const float*)d_in[15];
    const float* fc1_b    = (const float*)d_in[16];
    const float* fc2_w    = (const float*)d_in[17];
    const float* fc2_b    = (const float*)d_in[18];
    const float* outc_w   = (const float*)d_in[19];
    const float* bn_g     = (const float*)d_in[20];
    const float* bn_b     = (const float*)d_in[21];
    const float* bn_m     = (const float*)d_in[22];
    const float* bn_v     = (const float*)d_in[23];
    float* out            = (float*)d_out;

    char* ws = (char*)d_ws;
    // arena (bytes, all 256-aligned); xz region reused for m/mln/h1, xdbl for xo
    const size_t O_CH_F32 = 0;                           // 65536*64*4  = 16 MB
    const size_t O_CH_BF  = O_CH_F32 + 16777216;         // 65536*64*2  = 8 MB
    const size_t O_XZ     = O_CH_BF  + 8388608;          // 65536*256*4 = 64 MB
    const size_t O_M      = O_XZ;                        // reuse: 16 MB
    const size_t O_MLN    = O_XZ + 16777216;             // reuse: 8 MB
    const size_t O_H1     = O_XZ + 16777216 + 8388608;   // reuse: 32 MB
    const size_t O_XCBF   = O_XZ + 67108864;             // 65536*128*2 = 16 MB
    const size_t O_XDBL   = O_XCBF + 16777216;           // 65536*48*4  = 12 MB
    const size_t O_XO     = O_XDBL;                      // reuse: 8 MB (after scan+x_proj done)
    const size_t O_YZ     = O_XDBL + 12582912;           // 65536*128*2 = 16 MB
    const size_t O_W      = O_YZ + 16777216;             // weights
    unsigned short* w_inproj  = (unsigned short*)(ws + O_W);
    unsigned short* w_xproj   = (unsigned short*)(ws + O_W + 32768);
    unsigned short* w_outproj = (unsigned short*)(ws + O_W + 49152);
    unsigned short* w_fc1     = (unsigned short*)(ws + O_W + 65536);
    unsigned short* w_fc2     = (unsigned short*)(ws + O_W + 98304);
    unsigned short* w_outcT   = (unsigned short*)(ws + O_W + 131072);

    float*          ch_f32 = (float*)(ws + O_CH_F32);
    unsigned short* ch_bf  = (unsigned short*)(ws + O_CH_BF);
    float*          xz     = (float*)(ws + O_XZ);
    unsigned short* xc_bf  = (unsigned short*)(ws + O_XCBF);
    float*          xdbl   = (float*)(ws + O_XDBL);
    unsigned short* yz     = (unsigned short*)(ws + O_YZ);
    float*          m_f32  = (float*)(ws + O_M);
    unsigned short* mln    = (unsigned short*)(ws + O_MLN);
    unsigned short* h1     = (unsigned short*)(ws + O_H1);
    unsigned short* xo     = (unsigned short*)(ws + O_XO);

    // --- weight prep ---
    cvt_bf16_k<<<64, 256, 0, stream>>>(in_pw, w_inproj, 64 * 256);
    pad_xproj_k<<<24, 256, 0, stream>>>(x_pw, w_xproj);
    cvt_bf16_k<<<32, 256, 0, stream>>>(out_pw, w_outproj, 128 * 64);
    cvt_bf16_k<<<64, 256, 0, stream>>>(fc1_w, w_fc1, 64 * 256);
    cvt_bf16_k<<<64, 256, 0, stream>>>(fc2_w, w_fc2, 256 * 64);
    transpose_cvt_k<<<256, 256, 0, stream>>>(outc_w, w_outcT);

    // --- 1. input LN -> channel chunks ---
    ln_in_k<<<R2, 256, 0, stream>>>(x, ln_g, ln_b, ch_f32, ch_bf);

    // --- 2. in_proj: (65536,64)x(64,256) -> xz f32 ---
    gemm_bf16_k<EpiF32><<<dim3(16, 512), 256, 0, stream>>>(
        ch_bf, w_inproj, 64, 64, 256, EpiF32{xz, 256});

    // --- 3. depthwise causal conv + silu -> xc bf16 ---
    conv_silu_k<<<32768, 256, 0, stream>>>(xz, conv_w, conv_b, xc_bf);

    // --- 4. x_proj: (65536,128)x(128,48pad) -> xdbl f32 ---
    gemm_bf16_k<EpiF32><<<dim3(3, 512), 256, 0, stream>>>(
        xc_bf, w_xproj, 128, 128, 48, EpiF32{xdbl, 48});

    // --- 5. fused dt-proj + selective scan + gate -> yz bf16 ---
    scan_k<<<16, 128, 0, stream>>>(xdbl, xc_bf, xz, dt_pw, dt_pb, A_log, Dparam, yz);

    // --- 6. out_proj: (65536,128)x(128,64) -> m f32 ---
    gemm_bf16_k<EpiF32><<<dim3(4, 512), 256, 0, stream>>>(
        yz, w_outproj, 128, 128, 64, EpiF32{m_f32, 64});

    // --- 7. LN over D=64 -> mln bf16 ---
    ln_d64_k<<<MROWS / 8, 256, 0, stream>>>(m_f32, ln1_g, ln1_b, mln);

    // --- 8. fc1 + GELU: (65536,64)x(64,256) -> h1 bf16 ---
    gemm_bf16_k<EpiGelu><<<dim3(16, 512), 256, 0, stream>>>(
        mln, w_fc1, 64, 64, 256, EpiGelu{fc1_b, h1, 256});

    // --- 9. fc2 + bias + skip, scatter to xo (b,n,C) bf16 ---
    gemm_bf16_k<EpiSkip><<<dim3(4, 512), 256, 0, stream>>>(
        h1, w_fc2, 256, 256, 64, EpiSkip{fc2_b, ch_f32, skip_sc, xo});

    // --- 10. outc 1x1 conv + BN + SiLU -> d_out ---
    gemm_bf16_k<EpiBnSilu><<<dim3(16, 128), 256, 0, stream>>>(
        xo, w_outcT, 256, 256, 256, EpiBnSilu{bn_m, bn_v, bn_g, bn_b, out});
}
